// LookupAttention_69965017252058
// MI455X (gfx1250) — matrix operand order repaired and checked by reading, not compiled, not run on hardware
//
#include <hip/hip_runtime.h>

typedef __attribute__((ext_vector_type(16))) _Float16 v16h;
typedef __attribute__((ext_vector_type(8)))  float    v8f;
typedef __attribute__((ext_vector_type(4)))  unsigned int u32x4;
typedef __attribute__((ext_vector_type(8)))  int i32x8;
typedef __attribute__((ext_vector_type(4)))  int i32x4;

namespace {
constexpr int Nn  = 2048;
constexpr int Dd  = 64;
constexpr int Tt  = 8;
constexpr int Ss  = 256;
constexpr int Cc  = 8;
constexpr int BHh = 12;          // B*H
constexpr int WAVES = 8;         // 256 threads / 32
constexpr int TILES = Nn / 16;   // 128
}

// ---- Tensor Data Mover: 1-D contiguous copy descriptor (D# per ISA 08 §8.3/8.4) ----
// group0: [1:0]=count=1, [63:32]=lds_addr(bytes), [120:64]=global_addr, [127:126]=type=2
// group1: [17:16]=data_size(2 -> 4B), tensor_dim0, tensor_dim1=1, tile_dim0=n, dims 2+ unused
__device__ inline void tdm_desc_1d(unsigned lds_off_bytes, unsigned long long ga,
                                   unsigned n_dwords, u32x4* g0o, i32x8* g1o) {
  u32x4 g0;
  g0[0] = 1u;                                                 // count=1, user mode
  g0[1] = lds_off_bytes;                                      // lds_addr
  g0[2] = (unsigned)(ga & 0xffffffffu);                       // global_addr[31:0]
  g0[3] = (unsigned)((ga >> 32) & 0x01ffffffu) | (2u << 30);  // global_addr[56:32], type=2
  i32x8 g1;
  g1[0] = (int)(2u << 16);                                    // data_size = 4 bytes
  g1[1] = (int)((n_dwords & 0xffffu) << 16);                  // tensor_dim0[15:0]
  g1[2] = (int)(((n_dwords >> 16) & 0xffffu) | (1u << 16));   // tensor_dim0[31:16], tensor_dim1=1
  g1[3] = (int)((n_dwords & 0xffffu) << 16);                  // tile_dim0 = n
  g1[4] = 0;                                                  // tile_dim1=0, tile_dim2=0
  g1[5] = (int)n_dwords;                                      // tensor_dim0_stride lo
  g1[6] = 0;
  g1[7] = 0;
  *g0o = g0; *g1o = g1;
}

// Load one 16x32 f16 A-fragment (contiguous k order: k = 16*half + i) from an f32 row.
__device__ inline v16h load_a_f32(const float* __restrict__ row, int kk, int half) {
  v16h a;
  const int b0 = kk * 32 + half * 16;
  const int b1 = kk * 32 + half * 16 + 8;
#pragma unroll
  for (int i = 0; i < 8; ++i) a[i] = (_Float16)row[b0 + i];
#pragma unroll
  for (int i = 0; i < 8; ++i) a[8 + i] = (_Float16)row[b1 + i];
  return a;
}

// Analytic top-4 of the 2^8 bucket-weight products w(s) = prod_c (bit_c ? p_c : 1-p_c).
// Exact identities for p = sigmoid(x):
//   bit           = (x > 0)
//   max(p,1-p)    = 1 / (1 + exp(-|x|))       (one v_exp + one v_rcp, no division chain)
//   min/max ratio = exp(-|x|)
// Top-4 = { best, flip a, flip b, max(flip a+b, flip c) } for the 3 largest ratios.
// The 3-largest selection is a branchless select cascade (r>ra => r>rb => r>rc).
__device__ inline void top4_buckets(const float* __restrict__ sc,
                                    float* __restrict__ w_out,
                                    int* __restrict__ s_out) {
  float ratio[8];
  int s0 = 0;
  float wmax = 1.f;
#pragma unroll
  for (int c = 0; c < 8; ++c) {
    const float x = sc[c];
    const float r = __expf(-__builtin_fabsf(x));
    ratio[c] = r;
    s0 |= (x > 0.f) ? (1 << (7 - c)) : 0;
    wmax *= __builtin_amdgcn_rcpf(1.f + r);
  }
  float ra = -1.f, rb = -2.f, rc = -3.f;
  int ia = 0, ib = 0, ic = 0;
#pragma unroll
  for (int c = 0; c < 8; ++c) {
    const float r = ratio[c];
    const bool ga = r > ra, gb = r > rb, gc = r > rc;
    rc = gb ? rb : (gc ? r : rc);
    ic = gb ? ib : (gc ? c : ic);
    rb = ga ? ra : (gb ? r : rb);
    ib = ga ? ia : (gb ? c : ib);
    ra = ga ? r : ra;
    ia = ga ? c : ia;
  }
  const int ma = 1 << (7 - ia), mb = 1 << (7 - ib), mc = 1 << (7 - ic);
  w_out[0] = wmax;       s_out[0] = s0;
  w_out[1] = wmax * ra;  s_out[1] = s0 ^ ma;
  w_out[2] = wmax * rb;  s_out[2] = s0 ^ mb;
  const float rab = ra * rb;
  const bool abwins = rab >= rc;
  w_out[3] = wmax * (abwins ? rab : rc);
  s_out[3] = s0 ^ (abwins ? (ma ^ mb) : mc);
}

// ---------------- Phase 1: hash queries, scatter weighted values into LDS table ----
// grid.x = BH*T ; block = 256 (8 waves). Table slice (S*D f32 = 64 KB) lives in LDS.
__global__ void lookup_attn_scatter(const float* __restrict__ q,
                                    const float* __restrict__ v,
                                    const float* __restrict__ mask,
                                    const float* __restrict__ proj,
                                    float* __restrict__ tables) {
  extern __shared__ float lds[];
  const int tid  = threadIdx.x;
  const int lane = tid & 31;
  const int wave = tid >> 5;
  const int half = lane >> 4;
  const int col  = lane & 15;
  const int bh = blockIdx.x / Tt;   // head index (B==1)
  const int t  = blockIdx.x % Tt;

  float* table = lds;                                  // Ss*Dd floats
  float* wbuf  = lds + Ss * Dd + wave * 1408;
  float* sc    = wbuf;                                 // 16x16 score transpose
  float* vt    = wbuf + 256;                           // 16x64 raw v tile (TDM target)
  float* wts   = wbuf + 1280;                          // 16x4 top weights (mask folded in)
  int*   slots = (int*)(wbuf + 1344);                  // 16x4 top slots
  const unsigned vt_lds_off = (unsigned)((Ss * Dd + wave * 1408 + 256) * sizeof(float));

  for (int i = tid; i < Ss * Dd; i += blockDim.x) table[i] = 0.f;
  __syncthreads();

  // B fragment: proj[h,t,c,d] -> (K=d 64) x (N=c, cols 8..15 zero), 2 K-chunks
  const int   ccol = (col < Cc) ? col : 0;
  const float cscale = (col < Cc) ? 1.f : 0.f;
  const float* bsrc = proj + (((size_t)bh * Tt + t) * Cc + ccol) * Dd + half * 16;
  v16h b0, b1;
#pragma unroll
  for (int i = 0; i < 16; ++i) b0[i] = (_Float16)(bsrc[i] * cscale);
#pragma unroll
  for (int i = 0; i < 16; ++i) b1[i] = (_Float16)(bsrc[32 + i] * cscale);

  for (int it = 0; it < TILES / WAVES; ++it) {
    const int tile = wave + WAVES * it;
    const int row0 = tile * 16;
    const int tok  = row0 + col;

    // async: TDM-load this tile's contiguous 16x64 f32 value block into LDS,
    // overlapped with the WMMA hash + top-4 below.
    {
      u32x4 g0; i32x8 g1; i32x4 z4 = {}; i32x8 z8 = {};
      tdm_desc_1d(vt_lds_off,
                  (unsigned long long)(v + ((size_t)bh * Nn + row0) * Dd),
                  16 * Dd, &g0, &g1);
      __builtin_amdgcn_tensor_load_to_lds(g0, g1, z4, z4, z8, 0);
    }

    const float* qrow = q + ((size_t)bh * Nn + tok) * Dd;
    __builtin_prefetch(qrow + (size_t)WAVES * 16 * Dd, 0, 1);

    v16h a0 = load_a_f32(qrow, 0, half);
    v16h a1 = load_a_f32(qrow, 1, half);
    v8f acc = {};
    acc = __builtin_amdgcn_wmma_f32_16x16x32_f16(false, a0, false, b0, (short)0, acc, false, false);
    acc = __builtin_amdgcn_wmma_f32_16x16x32_f16(false, a1, false, b1, (short)0, acc, false, false);
#pragma unroll
    for (int r = 0; r < 8; ++r) sc[(r + 8 * half) * 16 + col] = acc[r];
    __syncthreads();

    if (lane < 16) {
      top4_buckets(&sc[lane * 16], &wts[lane * 4], &slots[lane * 4]);
      const float m = mask[row0 + lane];     // fold attention mask into weights
#pragma unroll
      for (int kk = 0; kk < 4; ++kk) wts[lane * 4 + kk] *= m;
    }
    __syncthreads();

    __builtin_amdgcn_s_wait_tensorcnt((short)0);   // vt tile resident

    // scatter: 16 tokens x 4 buckets, lanes split the D=64 payload
#pragma unroll 4
    for (int e = 0; e < 64; ++e) {
      const int tt = e >> 2;
      const float w  = wts[e];
      const int slot = slots[e];
      atomicAdd(&table[slot * Dd + lane],      w * vt[tt * Dd + lane]);
      atomicAdd(&table[slot * Dd + 32 + lane], w * vt[tt * Dd + 32 + lane]);
    }
    __syncthreads();
  }

  // single TDM store of the finished 64 KB table (contiguous 16384 dwords)
  if (wave == 0) {
    u32x4 g0; i32x8 g1; i32x4 z4 = {}; i32x8 z8 = {};
    tdm_desc_1d(0u,
                (unsigned long long)(tables + ((size_t)bh * Tt + t) * Ss * Dd),
                Ss * Dd, &g0, &g1);
    __builtin_amdgcn_tensor_store_from_lds(g0, g1, z4, z4, z8, 0);
    __builtin_amdgcn_s_wait_tensorcnt((short)0);
  }
}

// ---------------- Phase 2: hash keys, gather from tables (L2-resident) -------------
// grid = (BH, N/16/WAVES); block = 256; one 16-token tile per wave.
__global__ void lookup_attn_gather(const float* __restrict__ kq,
                                   const float* __restrict__ proj,
                                   const float* __restrict__ tables,
                                   float* __restrict__ out) {
  extern __shared__ float lds[];
  const int tid  = threadIdx.x;
  const int lane = tid & 31;
  const int wave = tid >> 5;
  const int half = lane >> 4;
  const int col  = lane & 15;
  const int bh = blockIdx.x;

  float* wbuf  = lds + wave * 2048;
  float* scw   = wbuf;                 // 16x64 score transpose
  float* wts   = wbuf + 1024;          // 16x8x4
  int*   slots = (int*)(wbuf + 1536);  // 16x8x4

  // B: 64(K=d) x 64(N = t*8+c): 4 column tiles x 2 K-chunks
  v16h bm[4][2];
#pragma unroll
  for (int nt = 0; nt < 4; ++nt) {
    const int cg = nt * 16 + col;
    const float* src = proj + (((size_t)bh * Tt + (cg >> 3)) * Cc + (cg & 7)) * Dd + half * 16;
#pragma unroll
    for (int kk = 0; kk < 2; ++kk)
#pragma unroll
      for (int i = 0; i < 16; ++i) bm[nt][kk][i] = (_Float16)src[kk * 32 + i];
  }

  const int tile = blockIdx.y * WAVES + wave;
  const int row0 = tile * 16;
  const float* krow = kq + ((size_t)bh * Nn + row0 + col) * Dd;
  v16h a0 = load_a_f32(krow, 0, half);
  v16h a1 = load_a_f32(krow, 1, half);

  v8f acc[4];
#pragma unroll
  for (int nt = 0; nt < 4; ++nt) {
    v8f c = {};
    c = __builtin_amdgcn_wmma_f32_16x16x32_f16(false, a0, false, bm[nt][0], (short)0, c, false, false);
    c = __builtin_amdgcn_wmma_f32_16x16x32_f16(false, a1, false, bm[nt][1], (short)0, c, false, false);
    acc[nt] = c;
  }
#pragma unroll
  for (int nt = 0; nt < 4; ++nt)
#pragma unroll
    for (int r = 0; r < 8; ++r)
      scw[(r + 8 * half) * 64 + nt * 16 + col] = acc[nt][r];
  __syncthreads();

  // all 32 lanes decode: lane handles token col, tables [half*4, half*4+4)
#pragma unroll
  for (int j = 0; j < 4; ++j) {
    const int tb8 = half * 4 + j;
    top4_buckets(&scw[col * 64 + tb8 * 8],
                 &wts[(col * Tt + tb8) * 4], &slots[(col * Tt + tb8) * 4]);
  }
  __syncthreads();

  const float* tb = tables + (size_t)bh * Tt * Ss * Dd;
  for (int tt = 0; tt < 16; ++tt) {
    float s0 = 0.f, s1 = 0.f;
#pragma unroll 4
    for (int e = 0; e < Tt * 4; ++e) {          // e = t*4 + k
      const int tidx = e >> 2;
      const float w  = wts[tt * 32 + e];
      const int slot = slots[tt * 32 + e];
      const float* rowp = tb + ((size_t)tidx * Ss + slot) * Dd;
      s0 += w * rowp[lane];
      s1 += w * rowp[32 + lane];
    }
    const int n2 = row0 + tt;
    out[((size_t)bh * Nn + n2) * Dd + lane]      = s0 * (1.f / Tt);
    out[((size_t)bh * Nn + n2) * Dd + 32 + lane] = s1 * (1.f / Tt);
  }
}

extern "C" void kernel_launch(void* const* d_in, const int* in_sizes, int n_in,
                              void* d_out, int out_size, void* d_ws, size_t ws_size,
                              hipStream_t stream) {
  (void)in_sizes; (void)n_in; (void)out_size; (void)ws_size;
  const float* q    = (const float*)d_in[0];
  const float* k    = (const float*)d_in[1];
  const float* v    = (const float*)d_in[2];
  const float* mask = (const float*)d_in[3];
  const float* proj = (const float*)d_in[4];
  float* out    = (float*)d_out;
  float* tables = (float*)d_ws;   // BH*T*S*D f32 = 6.3 MB

  const size_t lds1 = (size_t)(Ss * Dd + WAVES * 1408) * sizeof(float);  // ~108 KB
  const size_t lds2 = (size_t)(WAVES * 2048) * sizeof(float);            // 64 KB

  lookup_attn_scatter<<<dim3(BHh * Tt), dim3(256), lds1, stream>>>(q, v, mask, proj, tables);
  lookup_attn_gather<<<dim3(BHh, Nn / 16 / WAVES), dim3(256), lds2, stream>>>(k, proj, tables, out);
}
